// AdaptiveCNNBlock_75015898792468
// MI455X (gfx1250) — matrix operand order repaired and checked by reading, not compiled, not run on hardware
//
#include <hip/hip_runtime.h>
#include <hip/hip_bf16.h>
#include <stdint.h>

#ifndef __has_builtin
#define __has_builtin(x) 0
#endif

#define B_    32
#define T_    1000
#define F_    80
#define COUT_ 64
#define NAUX_ 128
#define KTAPS 9          // CIN*KH*KW = 1*3*3
#define TP_   500        // pooled T
#define FP_   40         // pooled F
#define TPB_  4          // pooled rows per block
#define NROW_ (2*TPB_+2) // halo-inclusive input rows per block (10)
#define NFRAG (2*TPB_*5) // B fragments per block: 8 conv rows x 5 f-tiles = 40

typedef __attribute__((ext_vector_type(16))) __bf16 v16bf;
typedef __attribute__((ext_vector_type(8)))  float  v8f;
typedef __attribute__((ext_vector_type(4)))  float  f32x4;
typedef __attribute__((ext_vector_type(4))) unsigned int u32x4;
typedef __attribute__((ext_vector_type(8)))  int    i32x8;
typedef __attribute__((ext_vector_type(4)))  int    i32x4;

// ---------------------------------------------------------------------------
// Kernel 1: hypernetwork (filters + biases) and length propagation.
// ---------------------------------------------------------------------------
__global__ __launch_bounds__(128)
void hyper_kernel(const float* __restrict__ aux,
                  const float* __restrict__ Wg, const float* __restrict__ bg,
                  const float* __restrict__ Wb, const float* __restrict__ bbv,
                  const int*   __restrict__ x_len,
                  float* __restrict__ filt_ws, float* __restrict__ bias_ws,
                  int*   __restrict__ l2_out) {
  __shared__ float s_aux[NAUX_];
  const int b   = blockIdx.x;
  const int tid = threadIdx.x;          // 128 threads
  s_aux[tid] = aux[b * NAUX_ + tid];
  __syncthreads();

  for (int row = tid; row < COUT_ * KTAPS; row += 128) {
    const float* w = Wg + (size_t)row * NAUX_;
    float acc = bg[row];
#pragma unroll 8
    for (int k = 0; k < NAUX_; ++k) acc = fmaf(s_aux[k], w[k], acc);
    filt_ws[b * COUT_ * KTAPS + row] = acc;
  }
  if (tid < COUT_) {
    const float* w = Wb + (size_t)tid * NAUX_;
    float acc = bbv[tid];
#pragma unroll 8
    for (int k = 0; k < NAUX_; ++k) acc = fmaf(s_aux[k], w[k], acc);
    bias_ws[b * COUT_ + tid] = acc;
  }
  if (tid == 0) {
    const int L = x_len[b];
    l2_out[b] = (L <= 1) ? 0 : (((L - 2) >> 1) + 1);
  }
}

// ---------------------------------------------------------------------------
// Kernel 2: fused conv (WMMA GEMM, K=9 padded to 32) + bias(in C) + relu +
// 2x2 maxpool + time mask. One block = (sample b, 4 pooled rows); 4 waves
// cover the 64 couts. x rows staged via TDM; B fragments built once in LDS
// and shared by all 4 waves; masked rows skip WMMA and zero-fill coalesced.
// ---------------------------------------------------------------------------
__global__ __launch_bounds__(128)
void adaptive_conv_wmma(const float* __restrict__ x,
                        const float* __restrict__ filt_ws,
                        const float* __restrict__ bias_ws,
                        const int*   __restrict__ x_len,
                        float* __restrict__ out) {
  __shared__ float s_x[NROW_ * F_];          // 10 halo-inclusive rows (3.2 KB)
  __shared__ v16bf s_frag[NFRAG * 16];       // 40 fragments x 16 cols x 32 B (20 KB)

  const int b    = blockIdx.x;
  const int tpb  = blockIdx.y;               // 0..124
  const int tbase = tpb * (2 * TPB_);        // first conv row of this block
  const int r_lo = (tbase - 1 < 0) ? 0 : (tbase - 1);
  const int r_hi = (tbase + 2 * TPB_ > T_ - 1) ? (T_ - 1) : (tbase + 2 * TPB_);
  const int nrows = r_hi - r_lo + 1;

  const int wave   = threadIdx.x >> 5;
  const int lane   = threadIdx.x & 31;
  const int half   = lane >> 4;              // 0: lanes 0-15 (K 0-15), 1: lanes 16-31
  const int lane16 = lane & 15;

  // ---- Stage x rows into LDS with the Tensor Data Mover (wave 0 issues) ----
#if __has_builtin(__builtin_amdgcn_tensor_load_to_lds)
  if (wave == 0) {
    const unsigned lds_addr = (unsigned)(uintptr_t)&s_x[0];
    const uint64_t ga = (uint64_t)(uintptr_t)(x + ((size_t)b * T_ + r_lo) * F_);
    u32x4 g0;
    g0[0] = 1u;                                            // count=1, user mode
    g0[1] = lds_addr;                                      // LDS dest
    g0[2] = (unsigned)(ga & 0xFFFFFFFFu);                  // global addr lo
    g0[3] = (unsigned)((ga >> 32) & 0x01FFFFFFu) | (2u << 30); // addr hi | type=2
    i32x8 g1;
    g1[0] = (int)(2u << 16);                               // data_size = 4 B
    g1[1] = (int)((unsigned)F_ << 16);                     // tensor_dim0 = 80
    g1[2] = (int)((unsigned)nrows << 16);                  // tensor_dim1 = nrows
    g1[3] = (int)((unsigned)F_ << 16);                     // tile_dim0 = 80
    g1[4] = (int)(unsigned)nrows;                          // tile_dim1 = nrows
    g1[5] = F_;                                            // tensor_dim0_stride = 80
    g1[6] = 0;
    g1[7] = 0;
    i32x4 gz4 = {0, 0, 0, 0};
    i32x8 gz8 = {0, 0, 0, 0, 0, 0, 0, 0};
    __builtin_amdgcn_tensor_load_to_lds(g0, g1, gz4, gz4, gz8, 0);
#if __has_builtin(__builtin_amdgcn_s_wait_tensorcnt)
    __builtin_amdgcn_s_wait_tensorcnt(0);
#else
    asm volatile("s_wait_tensorcnt 0" ::: "memory");
#endif
  }
#else
  for (int i = threadIdx.x; i < nrows * F_; i += 128)
    s_x[i] = x[((size_t)b * T_ + r_lo) * F_ + i];
#endif
  __syncthreads();

  // ---- Build the 40 shared B fragments, one column per task (640 tasks) ----
  // Fragment fi = localT*5 + ft ; column col -> conv position (tbase+localT,
  // ft*16+col). Element e (= K index for lanes 0-15) = tap k: kh=k/3, kw=k%3,
  // value = x[t+kh-1, n+kw-1], zero-padded at edges. K 9..15 zero.
#pragma unroll
  for (int j = 0; j < 5; ++j) {
    const int q      = threadIdx.x + 128 * j;   // 0..639
    const int fi     = q >> 4;
    const int col    = q & 15;
    const int localT = fi / 5;
    const int ft     = fi % 5;
    const int t = tbase + localT;
    const int n = ft * 16 + col;
    v16bf frag;
#pragma unroll
    for (int e = 0; e < 16; ++e) frag[e] = (__bf16)0.0f;
#pragma unroll
    for (int k = 0; k < KTAPS; ++k) {
      const int kh = k / 3, kw = k % 3;
      const int gt = t + kh - 1;
      const int gf = n + kw - 1;
      float v = 0.0f;
      if (gt >= 0 && gt < T_ && gf >= 0 && gf < F_)
        v = s_x[(gt - r_lo) * F_ + gf];
      frag[k] = (__bf16)v;
    }
    s_frag[q] = frag;
  }
  __syncthreads();

  // ---- A fragment (16 couts x K) ----
  // Lanes 0-15: row M=lane16, elems 0..8 = K 0..8 ; lanes 16-31: zero.
  const float* filt = filt_ws + ((size_t)b * COUT_ + wave * 16) * KTAPS;
  v16bf afrag;
#pragma unroll
  for (int e = 0; e < 16; ++e) afrag[e] = (__bf16)0.0f;
  if (half == 0) {
#pragma unroll
    for (int e = 0; e < 9; ++e) afrag[e] = (__bf16)filt[lane16 * KTAPS + e];
  }

  // Bias folded into WMMA C operand: D-tile row M = v + 8*half.
  v8f cbias;
#pragma unroll
  for (int v = 0; v < 8; ++v)
    cbias[v] = bias_ws[b * COUT_ + wave * 16 + half * 8 + v];

  const int L  = x_len[b];
  const int l2 = (L <= 1) ? 0 : (((L - 2) >> 1) + 1);

  // ---- 4 pooled rows x 5 f-tiles; 2 WMMAs per tile (conv rows 2tp, 2tp+1) --
#pragma unroll
  for (int ltp = 0; ltp < TPB_; ++ltp) {
    const int tp = tpb * TPB_ + ltp;
    const size_t obase =
        (((size_t)b * COUT_ + wave * 16 + half * 8) * TP_ + tp) * FP_;

    if (tp < l2) {                          // uniform branch per block row
#pragma unroll
      for (int ft = 0; ft < 5; ++ft) {
        const int fi0 = (2 * ltp) * 5 + ft;
        const int fi1 = (2 * ltp + 1) * 5 + ft;
        v16bf b0, b1;
#pragma unroll
        for (int e = 0; e < 16; ++e) { b0[e] = (__bf16)0.0f; b1[e] = (__bf16)0.0f; }
        if (half == 0) {                    // upper K-half lanes are all-zero
          b0 = s_frag[fi0 * 16 + lane16];
          b1 = s_frag[fi1 * 16 + lane16];
        }
        const v8f c0 = __builtin_amdgcn_wmma_f32_16x16x32_bf16(
            false, afrag, false, b0, (short)0, cbias, false, false);
        const v8f c1 = __builtin_amdgcn_wmma_f32_16x16x32_bf16(
            false, afrag, false, b1, (short)0, cbias, false, false);

        // relu + 2x2 maxpool (t-pair lane-local, f-pair via wave32 shuffle)
#pragma unroll
        for (int v = 0; v < 8; ++v) {
          float m = fmaxf(fmaxf(c0[v], c1[v]), 0.0f);
          m = fmaxf(m, __shfl_xor(m, 1, 32));
          if ((lane16 & 1) == 0) {
            const int f_out = (ft * 16 >> 1) + (lane16 >> 1);
            __builtin_nontemporal_store(
                m, out + obase + (size_t)v * (TP_ * FP_) + f_out);
          }
        }
      }
    } else {
      // masked row: coalesced float4 zero-fill of 16 couts x 40 f per wave
      const size_t zbase = (((size_t)b * COUT_ + wave * 16) * TP_ + tp) * FP_;
      const f32x4 z = {0.0f, 0.0f, 0.0f, 0.0f};
#pragma unroll
      for (int it = 0; it < 5; ++it) {
        const int s  = lane + 32 * it;      // 0..159 float4 slots
        const int cl = s / 10;              // cout_local
        const int fq = s - cl * 10;         // float4 index within row
        f32x4* p = (f32x4*)(out + zbase + (size_t)cl * (TP_ * FP_) + fq * 4);
        __builtin_nontemporal_store(z, p);
      }
    }
  }
}

// ---------------------------------------------------------------------------
extern "C" void kernel_launch(void* const* d_in, const int* in_sizes, int n_in,
                              void* d_out, int out_size, void* d_ws, size_t ws_size,
                              hipStream_t stream) {
  const float* x     = (const float*)d_in[0];
  const int*   x_len = (const int*)  d_in[1];
  const float* aux   = (const float*)d_in[2];
  const float* Wg    = (const float*)d_in[3];
  const float* bg    = (const float*)d_in[4];
  const float* Wb    = (const float*)d_in[5];
  const float* bbv   = (const float*)d_in[6];

  float* out     = (float*)d_out;
  float* filt_ws = (float*)d_ws;                           // 32*64*9 f32
  float* bias_ws = filt_ws + (size_t)B_ * COUT_ * KTAPS;   // 32*64 f32
  int*   l2_out  = (int*)(out + (size_t)B_ * COUT_ * TP_ * FP_);

  hyper_kernel<<<dim3(B_), dim3(128), 0, stream>>>(aux, Wg, bg, Wb, bbv, x_len,
                                                   filt_ws, bias_ws, l2_out);
  adaptive_conv_wmma<<<dim3(B_, TP_ / TPB_), dim3(128), 0, stream>>>(
      x, filt_ws, bias_ws, x_len, out);
}
